// CausalDepthWiseConv1d_77309411809
// MI455X (gfx1250) — matrix operand order repaired and checked by reading, not compiled, not run on hardware
//
#include <hip/hip_runtime.h>

typedef __attribute__((ext_vector_type(16))) _Float16 v16h;
typedef __attribute__((ext_vector_type(8)))  float    v8f;

namespace {
constexpr int BATCH = 4;
constexpr int L  = 4096;
constexpr int D  = 1024;
constexpr int BM = 128;       // M tile (rows of B*L)
constexpr int BN = 128;       // N tile (output channels)
constexpr int KC = 64;        // K chunk (input channels per stage)
constexpr int LDA = KC + 8;   // padded LDS stride in halves (144B, 16B-aligned rows)
}

struct U32x8 { uint4 a, b; };

// ---- pre-pass: fp32 w_pw -> fp16 copy in workspace (L2-resident thereafter) ----
__global__ __launch_bounds__(256) void cvt_wpw_kernel(const float* __restrict__ w,
                                                      _Float16* __restrict__ o) {
  int i = blockIdx.x * blockDim.x + threadIdx.x;          // D*D/4 threads
  float4 v = reinterpret_cast<const float4*>(w)[i];
  struct H4 { _Float16 a, b, c, d; };
  H4 h { (_Float16)v.x, (_Float16)v.y, (_Float16)v.z, (_Float16)v.w };
  reinterpret_cast<H4*>(o)[i] = h;
}

// ---- fused causal depthwise conv (K=3) + pointwise GEMM via WMMA ----
__global__ __launch_bounds__(256) void fused_dwpw_kernel(
    const float* __restrict__ x, const float* __restrict__ w_dw,
    const float* __restrict__ b_dw, const _Float16* __restrict__ wph,
    const float* __restrict__ b_pw, float* __restrict__ out) {
  __shared__ __align__(16) _Float16 Alds[BM * LDA];

  const int tid  = threadIdx.x;
  const int lane = tid & 31;
  const int wave = tid >> 5;
  const int wm   = wave & 1;    // 2 M groups of 64 rows
  const int wn   = wave >> 1;   // 4 N groups of 32 cols

  const int m0 = blockIdx.x * BM;        // row base; BM divides L so tile is in one batch
  const int n0 = blockIdx.y * BN;
  const int batch = m0 / L;
  const int l0 = m0 - batch * L;
  const float* xb = x + (size_t)batch * L * D;

  const int  ln16   = lane & 15;
  const bool hiHalf = lane >= 16;

  // accumulators, initialized with pointwise bias: C[m][n] = b_pw[n]
  v8f acc[4][2];
#pragma unroll
  for (int nf = 0; nf < 2; ++nf) {
    float bp = b_pw[n0 + wn * 32 + nf * 16 + ln16];
#pragma unroll
    for (int mf = 0; mf < 4; ++mf) {
      v8f c = {bp, bp, bp, bp, bp, bp, bp, bp};
      acc[mf][nf] = c;
    }
  }

  const int pc  = tid & (KC - 1);   // producer: channel within chunk
  const int pr0 = tid >> 6;         // producer: starting row (0..3)

  for (int kc = 0; kc < D; kc += KC) {
    // ---------- producer: depthwise conv for this K-chunk -> LDS fp16 ----------
    const int ch = kc + pc;
    const float w0 = w_dw[ch * 3 + 0];
    const float w1 = w_dw[ch * 3 + 1];
    const float w2 = w_dw[ch * 3 + 2];
    const float bd = b_dw[ch];
    __syncthreads();                       // previous chunk fully consumed
#pragma unroll 4
    for (int i = pr0; i < BM; i += 4) {    // rows are wave-uniform -> EXEC stays full
      const int l = l0 + i;
      const float xm2 = (l >= 2) ? xb[(size_t)(l - 2) * D + ch] : 0.0f;
      const float xm1 = (l >= 1) ? xb[(size_t)(l - 1) * D + ch] : 0.0f;
      const float x0  = xb[(size_t)l * D + ch];
      const float y = fmaf(xm2, w0, fmaf(xm1, w1, fmaf(x0, w2, bd)));
      Alds[i * LDA + pc] = (_Float16)y;
    }
    __syncthreads();

    // ---------- consumer: two K=32 WMMA steps over the chunk ----------
#pragma unroll
    for (int kk = 0; kk < KC; kk += 32) {
      // B fragments straight from global fp16 weights (B[k][n] = w_pw[n][k]):
      // lane n%16 = column, lane<16 -> K 0..15, lane>=16 -> K 16..31 (32B aligned)
      v16h bfrag[2];
#pragma unroll
      for (int nf = 0; nf < 2; ++nf) {
        const int col = n0 + wn * 32 + nf * 16 + ln16;
        const int kof = kc + kk + (hiHalf ? 16 : 0);
        bfrag[nf] = *reinterpret_cast<const v16h*>(wph + (size_t)col * D + kof);
      }
#pragma unroll
      for (int mf = 0; mf < 4; ++mf) {
        // A layout: lane<16 holds K 0..15, lane>=16 holds K 16..31
        const int row = wm * 64 + mf * 16 + ln16;
        const int ka  = kk + (hiHalf ? 16 : 0);
        const uint4 u0 = *reinterpret_cast<const uint4*>(&Alds[row * LDA + ka]);
        const uint4 u1 = *reinterpret_cast<const uint4*>(&Alds[row * LDA + ka + 8]);
        U32x8 uu{u0, u1};
        v16h afrag = __builtin_bit_cast(v16h, uu);
#pragma unroll
        for (int nf = 0; nf < 2; ++nf) {
          acc[mf][nf] = __builtin_amdgcn_wmma_f32_16x16x32_f16(
              false, afrag, false, bfrag[nf], (short)0, acc[mf][nf], false, false);
        }
      }
    }
  }

  // ---------- epilogue: C/D layout -> row-major fp32 output ----------
#pragma unroll
  for (int mf = 0; mf < 4; ++mf) {
    const int rowBase = m0 + wm * 64 + mf * 16 + (hiHalf ? 8 : 0);
#pragma unroll
    for (int nf = 0; nf < 2; ++nf) {
      const int col = n0 + wn * 32 + nf * 16 + ln16;
      float* op = out + (size_t)rowBase * D + col;
#pragma unroll
      for (int g = 0; g < 8; ++g) {
        op[(size_t)g * D] = acc[mf][nf][g];
      }
    }
  }
}

extern "C" void kernel_launch(void* const* d_in, const int* in_sizes, int n_in,
                              void* d_out, int out_size, void* d_ws, size_t ws_size,
                              hipStream_t stream) {
  (void)in_sizes; (void)n_in; (void)out_size; (void)ws_size;
  const float* x    = (const float*)d_in[0];
  const float* w_dw = (const float*)d_in[1];
  const float* b_dw = (const float*)d_in[2];
  const float* w_pw = (const float*)d_in[3];
  const float* b_pw = (const float*)d_in[4];
  float*    out = (float*)d_out;
  _Float16* wph = (_Float16*)d_ws;   // 2 MB fp16 weight copy

  cvt_wpw_kernel<<<(D * D / 4) / 256, 256, 0, stream>>>(w_pw, wph);

  dim3 grid((BATCH * L) / BM, D / BN);   // (128, 8) workgroups
  fused_dwpw_kernel<<<grid, 256, 0, stream>>>(x, w_dw, b_dw, wph, b_pw, out);
}
